// PConv_11467562680751
// MI455X (gfx1250) — matrix operand order repaired and checked by reading, not compiled, not run on hardware
//
#include <hip/hip_runtime.h>

// ---------------- types ----------------
typedef __attribute__((ext_vector_type(8)))  float          v8f;
typedef __attribute__((ext_vector_type(16))) __bf16         v16bf;
typedef __attribute__((ext_vector_type(16))) unsigned short v16u;

union BF16x16 { v16u u; v16bf b; uint4 q[2]; };

// ---------------- problem constants ----------------
#define Bn   4
#define Hn   192
#define Wn   192
#define HWn  (Hn * Wn)          // 36864
#define Cin  16
#define Cm   64
#define NP   (Bn * HWn)         // 147456 pixels
#define TPW  12                 // 16-pixel tiles per row
#define NTILES (Bn * Hn * TPW)  // 9216
#define PSTR 148                // padded per-pixel gz stride (dwords, 16B aligned)

// workspace layout
#define X1_FLOATS   (Bn * Cm * HWn)            // 9437184
#define LWB_BYTEOFF ((size_t)X1_FLOATS * 4)    // bf16 lw starts here (16B aligned)
#define P_BYTEOFF   (LWB_BYTEOFF + 65536 * 2)  // folded-param region

// offsets inside P (floats)
#define OFF_ELB 0      // [64]  folded final bias
#define OFF_EW0 64     // [64*16] folded conv0 weights
#define OFF_EB0 1088   // [64]
#define OFF_W1  1152   // [8*3]
#define OFF_B1  1176   // [8]
#define OFF_W2  1184   // [8*8]
#define OFF_B2  1248   // [8]
#define OFF_W3  1256   // [16*8]
#define OFF_B3  1384   // [16]

__device__ __forceinline__ unsigned short f2bf(float f) {
    unsigned int u = __float_as_uint(f);
    u += 0x7FFFu + ((u >> 16) & 1u);   // round-to-nearest-even
    return (unsigned short)(u >> 16);
}

// ---------------- kernel 0: fold BN into weights, lw -> bf16 ----------------
__global__ __launch_bounds__(256) void k_prep(
    const float* w0, const float* b0, const float* g0, const float* be0, const float* m0, const float* v0,
    const float* w1, const float* bb1, const float* g1, const float* be1, const float* m1, const float* v1,
    const float* w2, const float* bb2, const float* g2, const float* be2, const float* m2, const float* v2,
    const float* w3, const float* bb3, const float* g3, const float* be3, const float* m3, const float* v3,
    const float* lw, const float* lb, const float* lg, const float* lbe, const float* lm, const float* lv,
    float* P, unsigned short* lwb)
{
    const int t = threadIdx.x;
    const float eps = 1e-5f;
    if (t < 64) {
        float s = g0[t] * rsqrtf(v0[t] + eps);
        for (int c = 0; c < 16; ++c) P[OFF_EW0 + t * 16 + c] = w0[t * 16 + c] * s;
        P[OFF_EB0 + t] = be0[t] + (b0[t] - m0[t]) * s;
        float ls = lg[t] * rsqrtf(lv[t] + eps);
        P[OFF_ELB + t] = lbe[t] + (lb[t] - lm[t]) * ls;
    }
    if (t < 8) {
        float s = g1[t] * rsqrtf(v1[t] + eps);
        for (int c = 0; c < 3; ++c) P[OFF_W1 + t * 3 + c] = w1[t * 3 + c] * s;
        P[OFF_B1 + t] = be1[t] + (bb1[t] - m1[t]) * s;
    }
    if (t >= 64 && t < 72) {
        int r = t - 64;
        float s = g2[r] * rsqrtf(v2[r] + eps);
        for (int c = 0; c < 8; ++c) P[OFF_W2 + r * 8 + c] = w2[r * 8 + c] * s;
        P[OFF_B2 + r] = be2[r] + (bb2[r] - m2[r]) * s;
    }
    if (t >= 72 && t < 88) {
        int r = t - 72;
        float s = g3[r] * rsqrtf(v3[r] + eps);
        for (int c = 0; c < 8; ++c) P[OFF_W3 + r * 8 + c] = w3[r * 8 + c] * s;
        P[OFF_B3 + r] = be3[r] + (bb3[r] - m3[r]) * s;
    }
    // lw (64 x 1024) -> bf16 with row scale folded
    for (int i = t; i < 64 * 1024; i += 256) {
        int oo = i >> 10;
        float ls = lg[oo] * rsqrtf(lv[oo] + eps);
        lwb[i] = f2bf(lw[i] * ls);
    }
}

// ---------------- kernel 1: x1 = relu(BN(conv1x1 16->64)) ----------------
__global__ __launch_bounds__(256) void k_conv0(const float* __restrict__ x,
                                               const float* __restrict__ P,
                                               float* __restrict__ X1)
{
    int idx = blockIdx.x * 256 + threadIdx.x;
    if (idx >= NP) return;
    int b  = idx / HWn;
    int hw = idx % HWn;
    const float* EW0 = P + OFF_EW0;
    const float* EB0 = P + OFF_EB0;
    float xin[Cin];
#pragma unroll
    for (int c = 0; c < Cin; ++c)
        xin[c] = x[((size_t)(b * Cin + c)) * HWn + hw];
    for (int o = 0; o < Cm; ++o) {
        float a = EB0[o];
#pragma unroll
        for (int c = 0; c < Cin; ++c) a = fmaf(EW0[o * 16 + c], xin[c], a);
        X1[((size_t)(b * Cm + o)) * HWn + hw] = fmaxf(a, 0.f);
    }
}

// ---------------- kernel 2: fused WeightNet + patches + einsum + final linear ----------------
// One wave per 16-pixel tile: Y[64,16] = LWbf16[64,1024] @ T[1024,16] via
// 32 K-steps of v_wmma_f32_16x16x32_bf16 x 4 M-tiles. T is built on the fly
// from x1 3x3 patches (branch-free, mask-multiplied) and LDS-resident gz.
__global__ __launch_bounds__(256) void k_pconv_main(const float* __restrict__ gxyz,
                                                    const float* __restrict__ X1,
                                                    const unsigned short* __restrict__ LWB,
                                                    const float* __restrict__ P,
                                                    float* __restrict__ out)
{
    __shared__ __align__(16) float gzs[8][16 * PSTR];   // [wave][pix*PSTR + k*16 + o]

    const int lane = threadIdx.x & 31;
    const int wv   = threadIdx.x >> 5;
    const int tile = blockIdx.x * 8 + wv;
    if (tile >= NTILES) return;
    const int tw  = tile % TPW;
    const int h   = (tile / TPW) % Hn;
    const int b   = tile / (TPW * Hn);
    const int pix = lane & 15;
    const int w   = tw * 16 + pix;

    // ---- phase 0: WeightNet gz[9 k][16 o] per pixel (lanes 0..15) ----
    if (lane < 16) {
        const float* W1 = P + OFF_W1;  const float* B1 = P + OFF_B1;
        const float* W2 = P + OFF_W2;  const float* B2 = P + OFF_B2;
        const float* W3 = P + OFF_W3;  const float* B3 = P + OFF_B3;
        for (int k = 0; k < 9; ++k) {
            float i0 = gxyz[((((size_t)b * 3 + 0) * 9 + k) * Hn + h) * Wn + w];
            float i1 = gxyz[((((size_t)b * 3 + 1) * 9 + k) * Hn + h) * Wn + w];
            float i2 = gxyz[((((size_t)b * 3 + 2) * 9 + k) * Hn + h) * Wn + w];
            float l1[8], l2[8], l3[16];
#pragma unroll
            for (int o = 0; o < 8; ++o)
                l1[o] = fmaxf(fmaf(W1[o * 3 + 2], i2,
                              fmaf(W1[o * 3 + 1], i1,
                              fmaf(W1[o * 3 + 0], i0, B1[o]))), 0.f);
#pragma unroll
            for (int o = 0; o < 8; ++o) {
                float a = B2[o];
#pragma unroll
                for (int i = 0; i < 8; ++i) a = fmaf(W2[o * 8 + i], l1[i], a);
                l2[o] = fmaxf(a, 0.f);
            }
#pragma unroll
            for (int o = 0; o < 16; ++o) {
                float a = B3[o];
#pragma unroll
                for (int i = 0; i < 8; ++i) a = fmaf(W3[o * 8 + i], l2[i], a);
                l3[o] = fmaxf(a, 0.f);
            }
            float4* dst = (float4*)&gzs[wv][pix * PSTR + k * 16];
            dst[0] = make_float4(l3[0],  l3[1],  l3[2],  l3[3]);
            dst[1] = make_float4(l3[4],  l3[5],  l3[6],  l3[7]);
            dst[2] = make_float4(l3[8],  l3[9],  l3[10], l3[11]);
            dst[3] = make_float4(l3[12], l3[13], l3[14], l3[15]);
        }
    }
    // wave-local LDS producer/consumer: compiler's s_wait_dscnt suffices, no barrier.

    // ---- hoisted branch-free patch addressing: clamped offsets + {0,1} masks ----
    int   off[9];
    float msk[9];
#pragma unroll
    for (int ky = 0; ky < 3; ++ky)
#pragma unroll
        for (int kx = 0; kx < 3; ++kx) {
            int hh = h + ky - 1, ww = w + kx - 1;
            bool ok = (hh >= 0) & (hh < Hn) & (ww >= 0) & (ww < Wn);
            int hc = min(max(hh, 0), Hn - 1);
            int wc = min(max(ww, 0), Wn - 1);
            off[ky * 3 + kx] = hc * Wn + wc;
            msk[ky * 3 + kx] = ok ? 1.f : 0.f;
        }

    // ---- phase 1: K=1024 bf16 GEMM, 32 K-steps x 4 M-tiles ----
    v8f acc[4] = {};
    const int    cb  = lane >> 4;                 // channel parity within K-step
    const float* gzl = &gzs[wv][pix * PSTR];

    for (int s = 0; s < 32; ++s) {
        const float* x1c = X1 + ((size_t)(b * Cm + s * 2 + cb)) * HWn;
        // pull next channel's center line toward the WGP while WMMA runs
        __builtin_prefetch(x1c + 2 * HWn + off[4], 0, 1);

        float p[9];
#pragma unroll
        for (int k = 0; k < 9; ++k) p[k] = x1c[off[k]] * msk[k];

        // T[cb*16 + j, pix] for j=0..15 : vectorized LDS reads (4x ds_load_b128 per k)
        float t[16];
#pragma unroll
        for (int j = 0; j < 16; ++j) t[j] = 0.f;
#pragma unroll
        for (int k = 0; k < 9; ++k) {
            const float4* gr = (const float4*)(gzl + k * 16);
            float pk = p[k];
#pragma unroll
            for (int q = 0; q < 4; ++q) {
                float4 g = gr[q];
                t[q * 4 + 0] = fmaf(pk, g.x, t[q * 4 + 0]);
                t[q * 4 + 1] = fmaf(pk, g.y, t[q * 4 + 1]);
                t[q * 4 + 2] = fmaf(pk, g.z, t[q * 4 + 2]);
                t[q * 4 + 3] = fmaf(pk, g.w, t[q * 4 + 3]);
            }
        }
        // B fragment: lane half j -> K = cb*16 + j  (channel c, weightnet output j)
        BF16x16 bf;
#pragma unroll
        for (int j = 0; j < 16; ++j) bf.b[j] = (__bf16)t[j];

        // A fragments: M = mt*16 + (lane&15); halves 0-15 -> K=kb..kb+15
        const int kb = s * 32 + ((lane < 16) ? 0 : 16);
#pragma unroll
        for (int mt = 0; mt < 4; ++mt) {
            const unsigned short* ap = LWB + (size_t)(mt * 16 + pix) * 1024 + kb;
            BF16x16 af;
            af.q[0] = *(const uint4*)(ap);
            af.q[1] = *(const uint4*)(ap + 8);
            acc[mt] = __builtin_amdgcn_wmma_f32_16x16x32_bf16(
                false, af.b, false, bf.b, (short)0, acc[mt], false, false);
        }
    }

    // ---- epilogue: folded bias + relu; D layout: VGPR r -> M=r (lanes<16) / M=8+r ----
    const float* ELB = P + OFF_ELB;
    const int mrow = (lane < 16) ? 0 : 8;
#pragma unroll
    for (int mt = 0; mt < 4; ++mt)
#pragma unroll
        for (int r = 0; r < 8; ++r) {
            int M = mt * 16 + mrow + r;
            float y = acc[mt][r] + ELB[M];
            out[(((size_t)b * Cm + M) * Hn + h) * Wn + w] = fmaxf(y, 0.f);
        }
}

// ---------------- launcher ----------------
extern "C" void kernel_launch(void* const* d_in, const int* in_sizes, int n_in,
                              void* d_out, int out_size, void* d_ws, size_t ws_size,
                              hipStream_t stream) {
    const float* x     = (const float*)d_in[0];
    // d_in[1] = group_mask (unused by reference)
    const float* gxyz  = (const float*)d_in[2];
    const float* w0    = (const float*)d_in[3];
    const float* b0    = (const float*)d_in[4];
    const float* g0    = (const float*)d_in[5];
    const float* be0   = (const float*)d_in[6];
    const float* m0    = (const float*)d_in[7];
    const float* v0    = (const float*)d_in[8];
    const float* wnw0  = (const float*)d_in[9];
    const float* wnb0  = (const float*)d_in[10];
    const float* wng0  = (const float*)d_in[11];
    const float* wnbe0 = (const float*)d_in[12];
    const float* wnm0  = (const float*)d_in[13];
    const float* wnv0  = (const float*)d_in[14];
    const float* wnw1  = (const float*)d_in[15];
    const float* wnb1  = (const float*)d_in[16];
    const float* wng1  = (const float*)d_in[17];
    const float* wnbe1 = (const float*)d_in[18];
    const float* wnm1  = (const float*)d_in[19];
    const float* wnv1  = (const float*)d_in[20];
    const float* wnw2  = (const float*)d_in[21];
    const float* wnb2  = (const float*)d_in[22];
    const float* wng2  = (const float*)d_in[23];
    const float* wnbe2 = (const float*)d_in[24];
    const float* wnm2  = (const float*)d_in[25];
    const float* wnv2  = (const float*)d_in[26];
    const float* lw    = (const float*)d_in[27];
    const float* lb    = (const float*)d_in[28];
    const float* lg    = (const float*)d_in[29];
    const float* lbe   = (const float*)d_in[30];
    const float* lm    = (const float*)d_in[31];
    const float* lv    = (const float*)d_in[32];

    float*          X1  = (float*)d_ws;
    unsigned short* lwb = (unsigned short*)((char*)d_ws + LWB_BYTEOFF);
    float*          P   = (float*)((char*)d_ws + P_BYTEOFF);

    k_prep<<<1, 256, 0, stream>>>(w0, b0, g0, be0, m0, v0,
                                  wnw0, wnb0, wng0, wnbe0, wnm0, wnv0,
                                  wnw1, wnb1, wng1, wnbe1, wnm1, wnv1,
                                  wnw2, wnb2, wng2, wnbe2, wnm2, wnv2,
                                  lw, lb, lg, lbe, lm, lv, P, lwb);

    k_conv0<<<(NP + 255) / 256, 256, 0, stream>>>(x, P, X1);

    k_pconv_main<<<NTILES / 8, 256, 0, stream>>>(gxyz, X1, lwb, P, (float*)d_out);
}
